// NonLocalDenoising_45174466019391
// MI455X (gfx1250) — matrix operand order repaired and checked by reading, not compiled, not run on hardware
//
#include <hip/hip_runtime.h>
#include <math.h>

// Problem constants (fixed by the reference)
#define NN 4
#define CC 256
#define SS 4096   // H*W
#define EE 128    // embed dim

typedef __attribute__((ext_vector_type(16))) __bf16 v16bf;
typedef __attribute__((ext_vector_type(8)))  float  v8f;
typedef __attribute__((ext_vector_type(4)))  int    v4i;

union Frag { v16bf v; v4i q[2]; };

__device__ __forceinline__ v4i ld128(const __bf16* p) { return *(const v4i*)p; }

__device__ __forceinline__ v8f vzero() {
  v8f z = {0.f, 0.f, 0.f, 0.f, 0.f, 0.f, 0.f, 0.f};
  return z;
}

// Native base-2 exponential (v_exp_f32 is base-2 on AMD hardware)
__device__ __forceinline__ float fast_exp2(float x) {
#if __has_builtin(__builtin_amdgcn_exp2f)
  return __builtin_amdgcn_exp2f(x);
#else
  return exp2f(x);
#endif
}

__device__ __forceinline__ v8f wmma_bf16(const Frag& a, const Frag& b, v8f c) {
  // D = A(16x32 bf16) x B(32x16 bf16) + C(16x16 f32)
  return __builtin_amdgcn_wmma_f32_16x16x32_bf16(false, a.v, false, b.v,
                                                 (short)0, c, false, false);
}

// ---------------------------------------------------------------------------
// Kernel 0a: x[n][c][s] f32 -> xc[n][c][s] bf16 (V for P*V GEMM, B-frag layout)
//                           -> xT[n][s][c] bf16 (A-side for QK GEMM)
// ---------------------------------------------------------------------------
__global__ void k_prep_x(const float* __restrict__ x,
                         __bf16* __restrict__ xc, __bf16* __restrict__ xT) {
  __shared__ float tile[32][33];
  int n  = blockIdx.z;
  int c0 = blockIdx.y * 32;
  int s0 = blockIdx.x * 32;
  int tx = threadIdx.x & 31;
  int ty = threadIdx.x >> 5;  // 0..7

  const float* xp = x + ((size_t)n * CC + c0) * SS + s0;
  #pragma unroll
  for (int i = 0; i < 32; i += 8) {
    float v = xp[(size_t)(ty + i) * SS + tx];
    tile[ty + i][tx] = v;
    xc[((size_t)n * CC + c0 + ty + i) * SS + s0 + tx] = (__bf16)v;
  }
  __syncthreads();
  __bf16* op = xT + ((size_t)n * SS + s0) * CC + c0;
  #pragma unroll
  for (int i = 0; i < 32; i += 8)
    op[(size_t)(ty + i) * CC + tx] = (__bf16)tile[tx][ty + i];
}

// ---------------------------------------------------------------------------
// Kernel 0b: weights f32 -> bf16.
// theta gets 1/sqrt(E) (attention scale) AND log2(e) folded in, so the online
// softmax can use the native base-2 v_exp_f32 with no per-element multiply.
// softmax is invariant under a uniform change of base of the logits.
// ---------------------------------------------------------------------------
__global__ void k_prep_w(const float* __restrict__ tw, const float* __restrict__ pw,
                         const float* __restrict__ jw, __bf16* __restrict__ twb,
                         __bf16* __restrict__ pwb, __bf16* __restrict__ jwb) {
  int i = blockIdx.x * 256 + threadIdx.x;
  if (i < EE * CC) {
    twb[i] = (__bf16)(tw[i] * (0.0883883476483184405501313f *   // 1/sqrt(128)
                               1.4426950408889634073599247f));  // log2(e)
    pwb[i] = (__bf16)pw[i];
  }
  if (i < CC * CC) jwb[i] = (__bf16)jw[i];
}

// ---------------------------------------------------------------------------
// Kernel 1: Q[n][s][e] = scaled theta, K[n][t][e] = phi  (bf16, WMMA GEMM)
// One 16x16 tile per wave; both outputs share the A fragments.
// ---------------------------------------------------------------------------
__global__ void k_qk(const __bf16* __restrict__ xT, const __bf16* __restrict__ twb,
                     const __bf16* __restrict__ pwb, __bf16* __restrict__ Qs,
                     __bf16* __restrict__ Kt) {
  int wave = threadIdx.x >> 5, lane = threadIdx.x & 31;
  int lrow = lane & 15, lhalf = lane >> 4;
  int n   = blockIdx.y;
  int lin = blockIdx.x * 8 + wave;      // gridDim.x = 256 -> 2048 tiles per n
  int e0  = (lin & 7) * 16;             // 8 e-tiles
  int s0  = (lin >> 3) * 16;            // 256 s-tiles

  v8f cq = vzero(), ck = vzero();
  const __bf16* arow = xT + ((size_t)n * SS + s0 + lrow) * CC;
  const __bf16* tb   = twb + (size_t)(e0 + lrow) * CC;
  const __bf16* pb   = pwb + (size_t)(e0 + lrow) * CC;

  #pragma unroll
  for (int kc = 0; kc < 8; ++kc) {      // K = C = 256 in chunks of 32
    Frag a, b1, b2;
    a.q[0]  = ld128(arow + kc * 32 + lhalf * 8);
    a.q[1]  = ld128(arow + kc * 32 + 16 + lhalf * 8);
    b1.q[0] = ld128(tb + kc * 32 + lhalf * 8);
    b1.q[1] = ld128(tb + kc * 32 + 16 + lhalf * 8);
    b2.q[0] = ld128(pb + kc * 32 + lhalf * 8);
    b2.q[1] = ld128(pb + kc * 32 + 16 + lhalf * 8);
    cq = wmma_bf16(a, b1, cq);
    ck = wmma_bf16(a, b2, ck);
  }
  #pragma unroll
  for (int r = 0; r < 8; ++r) {
    size_t row = (size_t)n * SS + s0 + r + 8 * lhalf;
    Qs[row * EE + e0 + lrow] = (__bf16)cq[r];
    Kt[row * EE + e0 + lrow] = (__bf16)ck[r];
  }
}

// ---------------------------------------------------------------------------
// Kernel 2: flash attention. Each wave owns 16 rows of s; iterates t in 32-col
// steps: 8 score WMMAs, base-2 online softmax in f32, P->A-frag via
// wave-private LDS, 16 WMMAs accumulating O[16 x 256] in registers.
// 4 waves/WG -> 256 workgroups so the dominant kernel spreads across WGPs;
// K/V tiles still get 4x reuse through the read-only WGP$ (L2-resident: 3 MB
// per batch vs 192 MB L2). Writes y[n][s][c] bf16.
// ---------------------------------------------------------------------------
__global__ void __launch_bounds__(128, 1)
k_flash(const __bf16* __restrict__ Qs, const __bf16* __restrict__ Kt,
        const __bf16* __restrict__ xc, __bf16* __restrict__ Yb) {
  __shared__ float Pbuf[4][16][32];     // 8 KB, wave-private slices
  int wave = threadIdx.x >> 5, lane = threadIdx.x & 31;
  int lrow = lane & 15, lhalf = lane >> 4;
  int n  = blockIdx.y;
  int s0 = blockIdx.x * 64 + wave * 16;

  // Q A-fragments (held for whole kernel): 4 chunks of K(e)=32
  Frag qf[4];
  const __bf16* qrow = Qs + ((size_t)n * SS + s0 + lrow) * EE;
  #pragma unroll
  for (int ec = 0; ec < 4; ++ec) {
    qf[ec].q[0] = ld128(qrow + ec * 32 + lhalf * 8);
    qf[ec].q[1] = ld128(qrow + ec * 32 + 16 + lhalf * 8);
  }

  v8f o[16];                            // O tile: 16 rows x 256 cols f32
  #pragma unroll
  for (int i = 0; i < 16; ++i) o[i] = vzero();
  float m[8], l[8];
  #pragma unroll
  for (int r = 0; r < 8; ++r) { m[r] = -3.0e38f; l[r] = 0.f; }

  for (int t0 = 0; t0 < SS; t0 += 32) {
    // ---- scores (log2-domain): S = Q * K^T, two 16x16 tiles
    v8f sc[2];
    #pragma unroll
    for (int tc = 0; tc < 2; ++tc) {
      sc[tc] = vzero();
      const __bf16* krow = Kt + ((size_t)n * SS + t0 + tc * 16 + lrow) * EE;
      #pragma unroll
      for (int ec = 0; ec < 4; ++ec) {
        Frag kb;
        kb.q[0] = ld128(krow + ec * 32 + lhalf * 8);
        kb.q[1] = ld128(krow + ec * 32 + 16 + lhalf * 8);
        sc[tc] = wmma_bf16(qf[ec], kb, sc[tc]);
      }
    }
    // ---- base-2 online softmax: row stats via width-16 shfl reductions
    float corr[8];
    #pragma unroll
    for (int r = 0; r < 8; ++r) {
      float v = fmaxf(sc[0][r], sc[1][r]);
      #pragma unroll
      for (int off = 1; off < 16; off <<= 1) v = fmaxf(v, __shfl_xor(v, off, 16));
      float mn = fmaxf(m[r], v);
      corr[r] = fast_exp2(m[r] - mn);   // native v_exp_f32
      m[r] = mn;
    }
    #pragma unroll
    for (int tc = 0; tc < 2; ++tc)
      #pragma unroll
      for (int r = 0; r < 8; ++r)
        sc[tc][r] = fast_exp2(sc[tc][r] - m[r]);
    #pragma unroll
    for (int r = 0; r < 8; ++r) {
      float s = sc[0][r] + sc[1][r];
      #pragma unroll
      for (int off = 1; off < 16; off <<= 1) s += __shfl_xor(s, off, 16);
      l[r] = l[r] * corr[r] + s;
    }
    #pragma unroll
    for (int f = 0; f < 16; ++f)
      #pragma unroll
      for (int r = 0; r < 8; ++r)
        o[f][r] *= corr[r];

    // ---- P (C layout) -> LDS -> A-fragment layout (same wave, DS in-order)
    #pragma unroll
    for (int tc = 0; tc < 2; ++tc)
      #pragma unroll
      for (int r = 0; r < 8; ++r)
        Pbuf[wave][r + 8 * lhalf][tc * 16 + lrow] = sc[tc][r];

    Frag pa;
    #pragma unroll
    for (int j = 0; j < 4; ++j) {
      float2 p2 = *(const float2*)&Pbuf[wave][lrow][lhalf * 8 + 2 * j];
      pa.v[2 * j]     = (__bf16)p2.x;
      pa.v[2 * j + 1] = (__bf16)p2.y;
      float2 p3 = *(const float2*)&Pbuf[wave][lrow][16 + lhalf * 8 + 2 * j];
      pa.v[8 + 2 * j]     = (__bf16)p3.x;
      pa.v[8 + 2 * j + 1] = (__bf16)p3.y;
    }

    // ---- O += P(16x32) * V(32x256); V = xc[n][c][t], B-frag layout by const.
    const __bf16* vbase = xc + (size_t)n * CC * SS + (size_t)lrow * SS + t0 + lhalf * 8;
    #pragma unroll
    for (int cc = 0; cc < 16; ++cc) {
      Frag vb;
      const __bf16* p = vbase + (size_t)cc * 16 * SS;
      vb.q[0] = ld128(p);
      vb.q[1] = ld128(p + 16);
      o[cc] = wmma_bf16(pa, vb, o[cc]);
    }
  }

  // ---- normalize and store y[n][s][c] bf16
  #pragma unroll
  for (int r = 0; r < 8; ++r) l[r] = 1.0f / l[r];
  #pragma unroll
  for (int cc = 0; cc < 16; ++cc)
    #pragma unroll
    for (int r = 0; r < 8; ++r) {
      float val = o[cc][r] * l[r];
      Yb[((size_t)n * SS + s0 + r + 8 * lhalf) * CC + cc * 16 + lrow] = (__bf16)val;
    }
}

// ---------------------------------------------------------------------------
// Kernel 3: res = proj_w * y + x  -> out[n][c][s] f32 (fused residual)
// ---------------------------------------------------------------------------
__global__ void k_proj(const __bf16* __restrict__ Yb, const __bf16* __restrict__ jwb,
                       const float* __restrict__ x, float* __restrict__ out) {
  int wave = threadIdx.x >> 5, lane = threadIdx.x & 31;
  int lrow = lane & 15, lhalf = lane >> 4;
  int lin = blockIdx.x * 8 + wave;      // 16384 tiles total
  int s0  = (lin & 255) * 16;
  int o0  = ((lin >> 8) & 15) * 16;
  int n   = lin >> 12;

  v8f acc = vzero();
  const __bf16* arow = jwb + (size_t)(o0 + lrow) * CC;
  const __bf16* brow = Yb + ((size_t)n * SS + s0 + lrow) * CC;
  #pragma unroll
  for (int kc = 0; kc < 8; ++kc) {
    Frag a, b;
    a.q[0] = ld128(arow + kc * 32 + lhalf * 8);
    a.q[1] = ld128(arow + kc * 32 + 16 + lhalf * 8);
    b.q[0] = ld128(brow + kc * 32 + lhalf * 8);
    b.q[1] = ld128(brow + kc * 32 + 16 + lhalf * 8);
    acc = wmma_bf16(a, b, acc);
  }
  #pragma unroll
  for (int r = 0; r < 8; ++r) {
    size_t idx = ((size_t)n * CC + o0 + r + 8 * lhalf) * SS + s0 + lrow;
    out[idx] = x[idx] + acc[r];
  }
}

// ---------------------------------------------------------------------------
extern "C" void kernel_launch(void* const* d_in, const int* in_sizes, int n_in,
                              void* d_out, int out_size, void* d_ws, size_t ws_size,
                              hipStream_t stream) {
  (void)in_sizes; (void)n_in; (void)out_size; (void)ws_size;
  const float* x  = (const float*)d_in[0];
  const float* tw = (const float*)d_in[1];
  const float* pw = (const float*)d_in[2];
  const float* jw = (const float*)d_in[3];
  float* out = (float*)d_out;

  char* ws = (char*)d_ws;
  const size_t MB = 1u << 20;
  __bf16* xc  = (__bf16*)(ws);              //  8 MB: x as bf16, [n][c][s]
  __bf16* xT  = (__bf16*)(ws + 8 * MB);     //  8 MB: x^T bf16,  [n][s][c]
  __bf16* Qs  = (__bf16*)(ws + 16 * MB);    //  4 MB: scaled theta [n][s][e]
  __bf16* Kt  = (__bf16*)(ws + 20 * MB);    //  4 MB: phi          [n][t][e]
  __bf16* Yb  = (__bf16*)(ws + 24 * MB);    //  8 MB: attention out [n][s][c]
  __bf16* twb = (__bf16*)(ws + 32 * MB);    //  64 KB
  __bf16* pwb = twb + EE * CC;              //  64 KB
  __bf16* jwb = pwb + EE * CC;              // 128 KB

  k_prep_x<<<dim3(SS / 32, CC / 32, NN), 256, 0, stream>>>(x, xc, xT);
  k_prep_w<<<dim3((CC * CC + 255) / 256), 256, 0, stream>>>(tw, pw, jw, twb, pwb, jwb);
  k_qk   <<<dim3(256, NN), 256, 0, stream>>>(xT, twb, pwb, Qs, Kt);
  k_flash<<<dim3(SS / 64, NN), 128, 0, stream>>>(Qs, Kt, xc, Yb);
  k_proj <<<dim3(2048), 256, 0, stream>>>(Yb, jwb, x, out);
}
